// Network_3624952397869
// MI455X (gfx1250) — compile-verified, matrix operand order checked by reading, not run on hardware
//
#include <hip/hip_runtime.h>
#include <hip/hip_bf16.h>
#include <math.h>

typedef __bf16 bf16;
typedef __attribute__((ext_vector_type(16))) __bf16 v16bf;
typedef __attribute__((ext_vector_type(8)))  float  v8f;

union V16BF { v16bf v; bf16 e[16]; };
union V8F  { v8f  v; float e[8]; };

#define KB 8
#define KN 32
#define NPAIR (KB*KN*KN)   // 8192

// ---------------- device math helpers ----------------

__device__ inline double dfact_d(int n){ double f=1.0; for(int i=2;i<=n;++i) f*=(double)i; return f; }

__device__ double wigner3j(int j1,int j2,int j3,int m1,int m2,int m3){
  if (m1+m2+m3 != 0) return 0.0;
  int dj = (j1>j2)?(j1-j2):(j2-j1);
  if (j3 < dj || j3 > j1+j2) return 0.0;
  double pref = (((j1-j2-m3)&1)? -1.0 : 1.0) *
    sqrt(dfact_d(j1+j2-j3)*dfact_d(j1-j2+j3)*dfact_d(-j1+j2+j3)/dfact_d(j1+j2+j3+1)
         *dfact_d(j1+m1)*dfact_d(j1-m1)*dfact_d(j2+m2)*dfact_d(j2-m2)*dfact_d(j3+m3)*dfact_d(j3-m3));
  int kmin = 0;
  if (j2-j3-m1 > kmin) kmin = j2-j3-m1;
  if (j1-j3+m2 > kmin) kmin = j1-j3+m2;
  int kmax = j1+j2-j3;
  if (j1-m1 < kmax) kmax = j1-m1;
  if (j2+m2 < kmax) kmax = j2+m2;
  double s = 0.0;
  for (int k=kmin;k<=kmax;++k)
    s += ((k&1)?-1.0:1.0)/(dfact_d(k)*dfact_d(j1+j2-j3-k)*dfact_d(j1-m1-k)
         *dfact_d(j2+m2-k)*dfact_d(j3-j2+m1+k)*dfact_d(j3-j1-m2+k));
  return pref*s;
}

// complex->real SH change-of-basis matrix entry U[a][i] for degree l
__device__ void Uent(int l, int a, int i, double& re, double& im){
  re = 0.0; im = 0.0;
  const double s = 0.7071067811865476;
  if (a == l){ if (i == l) re = 1.0; return; }
  if (a > l){
    int m = a - l;
    if (i == l+m)      re = (m&1)? -s : s;
    else if (i == l-m) re = s;
  } else {
    int m = l - a;
    if (i == l-m)      im = s;
    else if (i == l+m) im = (m&1)? s : -s;   // -i*(-1)^m * s
  }
}

__device__ void c3elem(int lo,int li,int lf,int d1,int d2,
                       const double w3[7][7], const int k3[7][7],
                       int a,int b,int c, double& re, double& im){
  re = 0.0; im = 0.0;
  for (int i=0;i<d1;++i){
    double u1r,u1i; Uent(lo,a,i,u1r,u1i);
    if (u1r==0.0 && u1i==0.0) continue;
    for (int j=0;j<d2;++j){
      int kk = k3[i][j];
      if (kk < 0) continue;
      double wv = w3[i][j];
      if (wv == 0.0) continue;
      double u2r,u2i; Uent(li,b,j,u2r,u2i);
      if (u2r==0.0 && u2i==0.0) continue;
      double u3r,u3i; Uent(lf,c,kk,u3r,u3i);
      if (u3r==0.0 && u3i==0.0) continue;
      double tr = u1r*u2r - u1i*u2i;
      double ti = u1r*u2i + u1i*u2r;
      double pr = tr*u3r - ti*u3i;
      double pi = tr*u3i + ti*u3r;
      re += pr*wv; im += pi*wv;
    }
  }
}

__host__ __device__ constexpr int slot_index(int lo,int li,int lf){
  int s = 0;
  for (int o=0;o<4;++o) for (int i2=0;i2<4;++i2){
    int a0 = o>i2 ? o-i2 : i2-o;
    for (int f=a0; f<=o+i2; ++f){ if (o==lo && i2==li && f==lf) return s; ++s; }
  }
  return 0;
}

// branchless, overflow-safe softplus: max(y,0) + log(1 + exp(-|y|))
__device__ inline float dev_softplus(float y){
  return fmaxf(y, 0.f) + __logf(1.f + __expf(-fabsf(y)));
}

// WMMA 16-bit fragment K mapping: element e of the v16bf in lane-half `half`
__device__ inline int kmap(int e, int half){
  return (e < 8) ? (e + 8*half) : (16 + (e-8) + 8*half);
}

// ---------------- init: activation constants + Wigner 3j tensors ----------------

__global__ void k_init_tables(float* __restrict__ actc, float* __restrict__ c3j){
  int tid = threadIdx.x;
  if (tid >= 44 && tid <= 46){
    int which = tid - 44;             // 0: SSP_C, 1: SP_C, 2: SIG_C
    double sw=0.0, swf2=0.0;
    for (int ii=0; ii<=4000; ++ii){
      double x = -8.0 + 16.0*(double)ii/4000.0;
      double wgt = exp(-0.5*x*x);
      double splus = (x>0.0) ? 5.0*x + log1p(exp(-5.0*x)) : log1p(exp(5.0*x));
      double f;
      if (which==0)      f = splus/5.0 - log(2.0)/5.0;
      else if (which==1) f = splus/5.0;
      else               f = 1.0/(1.0+exp(-x));
      sw += wgt; swf2 += wgt*f*f;
    }
    actc[which] = (float)(1.0/sqrt(swf2/sw));
  }
  if (tid < 44){
    int lo=0, li=0, lf=0;
    { int cnt=0;
      for (int o=0;o<4;++o) for (int i2=0;i2<4;++i2){
        int a0 = o>i2 ? o-i2 : i2-o;
        for (int f=a0; f<=o+i2; ++f){ if (cnt==tid){lo=o;li=i2;lf=f;} cnt++; }
      } }
    int d1=2*lo+1, d2=2*li+1, d3=2*lf+1;
    double w3[7][7]; int k3[7][7];
    for (int i=0;i<d1;++i) for (int j=0;j<d2;++j){
      int m1=i-lo, m2=j-li, m3=-(m1+m2);
      if (m3 >= -lf && m3 <= lf){ w3[i][j]=wigner3j(lo,li,lf,m1,m2,m3); k3[i][j]=m3+lf; }
      else { w3[i][j]=0.0; k3[i][j]=-1; }
    }
    double sr=0.0, si=0.0;
    for (int a=0;a<d1;++a) for (int b=0;b<d2;++b) for (int c=0;c<d3;++c){
      double re,im; c3elem(lo,li,lf,d1,d2,w3,k3,a,b,c,re,im);
      sr += fabs(re); si += fabs(im);
    }
    bool useRe = (sr >= si);
    float* out = c3j + (size_t)tid*637;
    for (int a=0;a<d1;++a) for (int b=0;b<d2;++b) for (int c=0;c<d3;++c){
      double re,im; c3elem(lo,li,lf,d1,d2,w3,k3,a,b,c,re,im);
      out[(a*d2+b)*d3 + c] = (float)(useRe ? re : im);
    }
  }
}

// ---------------- geometry: spherical harmonics + radial basis ----------------

__global__ void k_geom(const float* __restrict__ geom, float* __restrict__ Yb,
                       bf16* __restrict__ basisb){
  int p = blockIdx.x*blockDim.x + threadIdx.x;
  if (p >= NPAIR) return;
  int z = p >> 10, rem = p & 1023, a = rem >> 5, b = rem & 31;
  const float* ga = geom + (size_t)(z*KN + a)*3;
  const float* gb = geom + (size_t)(z*KN + b)*3;
  float dx = ga[0]-gb[0], dy = ga[1]-gb[1], dz = ga[2]-gb[2];
  if (a == b){ dx += 0.3f; dy += 0.4f; dz += 0.866f; }
  float r  = sqrtf(dx*dx + dy*dy + dz*dz + 1e-12f);
  float ct = dz / r;
  float st = sqrtf(fmaxf(1.f - ct*ct, 1e-12f));
  float phi = atan2f(dy, dx);
  float P[7][7];
  P[0][0] = 1.f;
  for (int m=1;m<=6;++m) P[m][m] = (float)(2*m-1)*st*P[m-1][m-1];
  for (int m=0;m<=5;++m) P[m+1][m] = (float)(2*m+1)*ct*P[m][m];
  for (int m=0;m<=6;++m)
    for (int l=m+2;l<=6;++l)
      P[l][m] = ((float)(2*l-1)*ct*P[l-1][m] - (float)(l+m-1)*P[l-2][m]) / (float)(l-m);
  float* Yp = Yb + (size_t)p*49;
  const float SQ2 = 1.41421356237f;
  const double FOURPI = 12.566370614359172;
  for (int l=0;l<=6;++l){
    for (int m=-l;m<=l;++m){
      int am = (m<0) ? -m : m;
      float Kc = (float)sqrt((2.0*l+1.0)/FOURPI * dfact_d(l-am)/dfact_d(l+am));
      float v;
      if (m == 0)      v = Kc*P[l][0];
      else if (m > 0)  v = SQ2*Kc*cosf((float)m*phi)*P[l][m];
      else             v = SQ2*Kc*sinf((float)am*phi)*P[l][am];
      Yp[l*l + (m+l)] = v;
    }
  }
  const float step = 5.0f/31.0f;
  bf16* bp = basisb + (size_t)p*32;
  for (int i=0;i<32;++i){
    float cc = (float)i*step;
    float d  = (r - cc)/step;
    float h  = 0.f;
    if (fabsf(d) < 1.f){ float cx = cosf(1.57079632679f*d); h = cx*cx; }
    bp[i] = (bf16)h;
  }
}

// ---------------- misc small kernels ----------------

__global__ void k_embed(const int* __restrict__ feats, const float* __restrict__ emb,
                        float* __restrict__ featA){
  int idx = blockIdx.x*blockDim.x + threadIdx.x;
  if (idx >= KB*KN*32) return;
  int d = idx & 31, n = idx >> 5;
  featA[idx] = emb[feats[n]*32 + d];
}

__global__ void k_prescale(const float* __restrict__ in, float* __restrict__ out,
                           const float* __restrict__ mask, int din){
  int idx = blockIdx.x*blockDim.x + threadIdx.x;
  int total = KB*KN*din;
  if (idx >= total) return;
  int z = idx / (KN*din);
  float na = 0.f;
  for (int n=0;n<KN;++n) na += mask[z*KN + n];
  out[idx] = in[idx] * rsqrtf(na);
}

__global__ void k_f2bf(const float* __restrict__ src, bf16* __restrict__ dst, int n){
  int i = blockIdx.x*blockDim.x + threadIdx.x;
  if (i < n) dst[i] = (bf16)src[i];
}

// pre-swizzle a [Kdim x 128] f32 weight matrix into WMMA B-fragment order (bf16):
// dst[((nt*(Kdim/32) + kb)*32 + lane)*16 + e] = W[(kb*32 + kmap(e,lane>>4))*128 + nt*16 + (lane&15)]
__global__ void k_wswiz(const float* __restrict__ src, bf16* __restrict__ dst, int Kdim){
  int idx = blockIdx.x*blockDim.x + threadIdx.x;
  int nkb = Kdim >> 5;
  int total = 8*nkb*32*16;
  if (idx >= total) return;
  int e    = idx & 15;
  int lane = (idx >> 4) & 31;
  int kb   = (idx >> 9) % nkb;
  int nt   = (idx >> 9) / nkb;
  int k = kb*32 + kmap(e, lane >> 4);
  int n = nt*16 + (lane & 15);
  dst[idx] = (bf16)src[k*128 + n];
}

// ---------------- WMMA GEMM (one wave = 16-row stripe x full N=128) ----------------
// A: [M x KK] bf16 row-major; Bs: pre-swizzled fragments; out = ssp(A@B) as bf16.

template<int KK>
__global__ void __launch_bounds__(32)
k_gemm_ssp(const bf16* __restrict__ A, const bf16* __restrict__ Bs,
           bf16* __restrict__ Cm, const float* __restrict__ actc){
  constexpr int NN  = 128;
  constexpr int NKB = KK/32;
  const int mt   = blockIdx.x;
  const int lane = threadIdx.x;
  const int half = lane >> 4;
  const int rc   = lane & 15;
  const int row  = mt*16 + rc;

  V16BF av[NKB];
  #pragma unroll
  for (int kb=0; kb<NKB; ++kb){
    const bf16* ar = A + (size_t)row*KK + kb*32;
    ((uint4*)&av[kb])[0] = *(const uint4*)(ar + 8*half);
    ((uint4*)&av[kb])[1] = *(const uint4*)(ar + 16 + 8*half);
  }
  const float SSPc = actc[0];
  #pragma unroll
  for (int nt=0; nt<8; ++nt){
    const int col = nt*16 + rc;
    V8F c;
    #pragma unroll
    for (int r=0;r<8;++r) c.e[r] = 0.f;
    #pragma unroll
    for (int kb=0; kb<NKB; ++kb){
      const bf16* bp = Bs + ((size_t)(nt*NKB + kb)*32 + lane)*16;
      V16BF bv;
      ((uint4*)&bv)[0] = ((const uint4*)bp)[0];
      ((uint4*)&bv)[1] = ((const uint4*)bp)[1];
      c.v = __builtin_amdgcn_wmma_f32_16x16x32_bf16(false, av[kb].v, false, bv.v,
                                                    (short)0, c.v, false, false);
    }
    #pragma unroll
    for (int r=0;r<8;++r){
      int m = r + 8*half;
      float x = SSPc*(dev_softplus(5.f*c.e[r]) - 0.69314718056f)*0.2f;
      Cm[(size_t)(mt*16 + m)*NN + col] = (bf16)x;
    }
  }
}

// ---------------- fused (H2 @ W2) + equivariant tensor-product convolution ----------------

template<int LAYER> struct Cfg;
template<> struct Cfg<0>{
  static constexpr int NOUT=5, NIN=1, DIN=32, DOUT=176;
  static constexpr int MO[5]={32,16,8,4,28};
  static constexpr int LOa[5]={0,1,2,3,0};
  static constexpr int OOFF[5]={0,32,80,120,148};
  static constexpr int MI[1]={32};
  static constexpr int LIa[1]={0};
  static constexpr int IOFF[1]={0};
};
template<> struct Cfg<1>{
  static constexpr int NOUT=5, NIN=4, DIN=148, DOUT=176;
  static constexpr int MO[5]={32,16,8,4,28};
  static constexpr int LOa[5]={0,1,2,3,0};
  static constexpr int OOFF[5]={0,32,80,120,148};
  static constexpr int MI[4]={32,16,8,4};
  static constexpr int LIa[4]={0,1,2,3};
  static constexpr int IOFF[4]={0,32,80,120};
};
template<> struct Cfg<2>{
  static constexpr int NOUT=1, NIN=4, DIN=148, DOUT=1;
  static constexpr int MO[1]={1};
  static constexpr int LOa[1]={0};
  static constexpr int OOFF[1]={0};
  static constexpr int MI[4]={32,16,8,4};
  static constexpr int LIa[4]={0,1,2,3};
  static constexpr int IOFF[4]={0,32,80,120};
};

template<int LAYER>
__global__ void __launch_bounds__(128)
k_conv(const bf16* __restrict__ H2, const bf16* __restrict__ W2b,
       const float* __restrict__ featS, const float* __restrict__ Yb,
       const float* __restrict__ c3j, float* __restrict__ convout, int nwTot){
  using C = Cfg<LAYER>;
  __shared__ float s_rw[16*512];          // 16 neighbors x up-to-512 radial weights (32 KB)
  __shared__ float s_feat[KN*C::DIN];     // all neighbor features of this molecule
  __shared__ float s_y[KN*49];            // Y(l<=6) for all neighbors of atom a
  __shared__ float s_acc[192];
  const int z = blockIdx.x >> 5;
  const int a = blockIdx.x & 31;
  const int lane = threadIdx.x & 31;
  const int wv   = threadIdx.x >> 5;
  const int half = lane >> 4;

  // --- async global->LDS staging of features and spherical harmonics (ASYNCcnt) ---
  {
    const float* gf = featS + (size_t)z*KN*C::DIN;
    const float* gy = Yb + (size_t)((z*KN + a)*KN)*49;
    const int nf = KN*C::DIN;             // multiple of 4 dwords, 16B aligned
    for (int t4 = threadIdx.x*4; t4 < nf; t4 += 128*4){
      unsigned lds_off = (unsigned)(unsigned long long)(&s_feat[t4]);
      const float* src = gf + t4;
      asm volatile("global_load_async_to_lds_b128 %0, %1, off"
                   :: "v"(lds_off), "v"(src) : "memory");
    }
    for (int t = threadIdx.x; t < KN*49; t += 128){
      unsigned lds_off = (unsigned)(unsigned long long)(&s_y[t]);
      const float* src = gy + t;
      asm volatile("global_load_async_to_lds_b32 %0, %1, off"
                   :: "v"(lds_off), "v"(src) : "memory");
    }
    asm volatile("s_wait_asynccnt 0" ::: "memory");
  }
  for (int t=threadIdx.x; t<C::DOUT; t+=128) s_acc[t] = 0.f;
  __syncthreads();

  for (int bg=0; bg<2; ++bg){
    const int basePair = (z*KN + a)*KN + bg*16;
    // A fragments: H2 rows for the 16 neighbors of this group, K=128
    V16BF afr[4];
    const bf16* Arow = H2 + (size_t)(basePair + (lane & 15)) * 128;
    #pragma unroll
    for (int kb=0; kb<4; ++kb){
      ((uint4*)&afr[kb])[0] = *(const uint4*)(Arow + kb*32 + 8*half);
      ((uint4*)&afr[kb])[1] = *(const uint4*)(Arow + kb*32 + 16 + 8*half);
    }
    int w_off = 0;
    for (int ob=0; ob<C::NOUT; ++ob){
      const int mo = C::MO[ob], lo = C::LOa[ob], ooff = C::OOFF[ob];
      for (int ib=0; ib<C::NIN; ++ib){
        const int mi = C::MI[ib], li = C::LIa[ib], ioff = C::IOFF[ib];
        const int lfmin = (lo>li) ? lo-li : li-lo;
        const int lfmax = lo+li;
        const float norm = rsqrtf((float)(mi*(lfmax-lfmin+1)));
        for (int lf=lfmin; lf<=lfmax; ++lf){
          const int umax = ((512/mi) < mo) ? (512/mi) : mo;
          for (int u0=0; u0<mo; u0+=umax){
            const int ucnt = ((mo-u0) < umax) ? (mo-u0) : umax;
            const int cols = ucnt*mi;
            const int colBase = w_off + u0*mi;
            const int colEnd  = colBase + cols;
            const int ntiles = (cols + 15) >> 4;
            // --- WMMA: Rw[16 x cols] = H2_block @ W2[:, colBase:colEnd] ---
            for (int t=wv; t<ntiles; t+=4){
              V8F c;
              #pragma unroll
              for (int r=0;r<8;++r) c.e[r] = 0.f;
              const int gcol = colBase + t*16 + (lane & 15);
              const bool ok = (gcol < colEnd);
              #pragma unroll
              for (int kb=0; kb<4; ++kb){
                V16BF bfr;
                #pragma unroll
                for (int e=0;e<16;++e){
                  int k = kb*32 + kmap(e, half);
                  bfr.e[e] = ok ? W2b[(size_t)k*nwTot + gcol] : (bf16)0.f;
                }
                c.v = __builtin_amdgcn_wmma_f32_16x16x32_bf16(false, afr[kb].v, false, bfr.v,
                                                              (short)0, c.v, false, false);
              }
              const int lcol = t*16 + (lane & 15);
              if (lcol < cols){
                #pragma unroll
                for (int r=0;r<8;++r){
                  int m = r + 8*half;
                  s_rw[m*512 + lcol] = c.e[r];
                }
              }
            }
            __syncthreads();
            // --- contraction: acc[u,i] += norm * sum_{b,v,j,m} Rw * C3J*Y * f ---
            const int d1 = 2*lo+1, d2 = 2*li+1, d3 = 2*lf+1;
            const int entries = ucnt*d1;
            if (threadIdx.x < entries){
              const int uu = threadIdx.x / d1;
              const int i  = threadIdx.x % d1;
              const float* c3 = c3j + (size_t)slot_index(lo,li,lf)*637 + i*d2*d3;
              float sum = 0.f;
              for (int bl=0; bl<16; ++bl){
                const int b = bg*16 + bl;
                if (b == a) continue;   // (1 - eye) mask on Rw
                const float* Yp = s_y + b*49 + lf*lf;
                float q[7];
                for (int j=0;j<d2;++j){
                  float s = 0.f;
                  for (int m=0;m<d3;++m) s += c3[j*d3 + m]*Yp[m];
                  q[j] = s;
                }
                const float* fp  = s_feat + b*C::DIN + ioff;
                const float* rwp = s_rw + bl*512 + uu*mi;
                for (int vvi=0; vvi<mi; ++vvi){
                  const float* fv = fp + vvi*d2;
                  float dot = 0.f;
                  for (int j=0;j<d2;++j) dot += q[j]*fv[j];
                  sum += rwp[vvi]*dot;
                }
              }
              s_acc[ooff + (u0+uu)*d1 + i] += norm*sum;
            }
            __syncthreads();
          }
          w_off += mo*mi;
        }
      }
    }
  }
  float* o = convout + (size_t)(z*KN + a)*C::DOUT;
  for (int t=threadIdx.x; t<C::DOUT; t+=128) o[t] = s_acc[t];
}

// ---------------- gate / output ----------------

__global__ void k_gate(const float* __restrict__ conv, const float* __restrict__ mask,
                       const float* __restrict__ actc, float* __restrict__ feat){
  int t = blockIdx.x*blockDim.x + threadIdx.x;
  if (t >= KB*KN) return;
  const float* in = conv + (size_t)t*176;
  float* o = feat + (size_t)t*148;
  float mk = mask[t];
  float SPc = actc[1], SIGc = actc[2];
  for (int d=0; d<32; ++d)
    o[d] = SPc*dev_softplus(5.f*in[d])*0.2f * mk;
  float g[28];
  for (int k=0;k<28;++k) g[k] = SIGc/(1.f + __expf(-in[148+k]));
  int go = 0, off = 32;
  const int gm[3] = {16,8,4};
  const int gl[3] = {1,2,3};
  for (int bl=0; bl<3; ++bl){
    int m = gm[bl], dd = 2*gl[bl]+1;
    for (int u=0; u<m; ++u)
      for (int j=0; j<dd; ++j)
        o[off + u*dd + j] = in[off + u*dd + j]*g[go+u]*mk;
    go += m; off += m*dd;
  }
}

__global__ void k_final(const float* __restrict__ conv, const float* __restrict__ mask,
                        const float* __restrict__ actc, float* __restrict__ out){
  int t = blockIdx.x*blockDim.x + threadIdx.x;
  if (t >= KB*KN) return;
  out[t] = actc[1]*dev_softplus(5.f*conv[t])*0.2f * mask[t];
}

// ---------------- launch ----------------

extern "C" void kernel_launch(void* const* d_in, const int* in_sizes, int n_in,
                              void* d_out, int out_size, void* d_ws, size_t ws_size,
                              hipStream_t stream){
  const int*   features = (const int*)d_in[0];
  const float* geometry = (const float*)d_in[1];
  const float* mask     = (const float*)d_in[2];
  const float* embed    = (const float*)d_in[3];
  const float* W0[3] = {(const float*)d_in[4], (const float*)d_in[7],  (const float*)d_in[10]};
  const float* W1[3] = {(const float*)d_in[5], (const float*)d_in[8],  (const float*)d_in[11]};
  const float* W2[3] = {(const float*)d_in[6], (const float*)d_in[9],  (const float*)d_in[12]};
  const int NW[3]  = {2816, 7168, 60};
  const int DIN[3] = {32, 148, 148};
  (void)in_sizes; (void)n_in; (void)out_size; (void)ws_size;

  char* w = (char*)d_ws;
  size_t off = 0;
  auto take = [&](size_t bytes)->char*{
    char* p = w + off; off = (off + bytes + 255) & ~(size_t)255; return p;
  };
  float* actc    = (float*)take(16);
  float* c3j     = (float*)take((size_t)44*637*4);
  float* Yb      = (float*)take((size_t)NPAIR*49*4);
  bf16*  basisb  = (bf16*) take((size_t)NPAIR*32*2);
  bf16*  h1b     = (bf16*) take((size_t)NPAIR*128*2);
  bf16*  h2b     = (bf16*) take((size_t)NPAIR*128*2);
  bf16*  w0s     = (bf16*) take((size_t)32*128*2);     // swizzled fragments
  bf16*  w1s     = (bf16*) take((size_t)128*128*2);    // swizzled fragments
  bf16*  w2b     = (bf16*) take((size_t)128*7168*2);
  float* featA   = (float*)take((size_t)KB*KN*148*4);
  float* featS   = (float*)take((size_t)KB*KN*148*4);
  float* convout = (float*)take((size_t)KB*KN*176*4);

  k_init_tables<<<1, 64, 0, stream>>>(actc, c3j);
  k_geom<<<NPAIR/128, 128, 0, stream>>>(geometry, Yb, basisb);
  k_embed<<<(KB*KN*32)/256, 256, 0, stream>>>(features, embed, featA);

  for (int L=0; L<3; ++L){
    // weights: W0/W1 converted + swizzled into WMMA fragment order, W2 plain bf16
    k_wswiz<<<(32*128 + 255)/256, 256, 0, stream>>>(W0[L], w0s, 32);
    k_wswiz<<<(128*128 + 255)/256, 256, 0, stream>>>(W1[L], w1s, 128);
    k_f2bf<<<(128*NW[L] + 255)/256, 256, 0, stream>>>(W2[L], w2b, 128*NW[L]);

    // radial MLP: basis(32) -> 128 -> 128, ssp fused, bf16 activations
    k_gemm_ssp<32><<<NPAIR/16, 32, 0, stream>>>(basisb, w0s, h1b, actc);
    k_gemm_ssp<128><<<NPAIR/16, 32, 0, stream>>>(h1b, w1s, h2b, actc);

    int totS = KB*KN*DIN[L];
    k_prescale<<<(totS + 255)/256, 256, 0, stream>>>(featA, featS, mask, DIN[L]);

    if (L == 0)
      k_conv<0><<<KB*KN, 128, 0, stream>>>(h2b, w2b, featS, Yb, c3j, convout, NW[L]);
    else if (L == 1)
      k_conv<1><<<KB*KN, 128, 0, stream>>>(h2b, w2b, featS, Yb, c3j, convout, NW[L]);
    else
      k_conv<2><<<KB*KN, 128, 0, stream>>>(h2b, w2b, featS, Yb, c3j, convout, NW[L]);

    if (L < 2)
      k_gate<<<1, 256, 0, stream>>>(convout, mask, actc, featA);
    else
      k_final<<<1, 256, 0, stream>>>(convout, mask, actc, (float*)d_out);
  }
}
